// Quantizer_67671504716402
// MI455X (gfx1250) — matrix operand order repaired and checked by reading, not compiled, not run on hardware
//
#include <hip/hip_runtime.h>

// ---------------------------------------------------------------------------
// VQ quantizer for MI455X (gfx1250, wave32).
//  - z @ codebook^T via v_wmma_f32_16x16x32_bf16 with bf16 hi/lo split of both
//    operands (3 WMMAs -> near-fp32 dot product accuracy).
//  - Codebook pre-split to bf16 hi/lo in workspace; tiles streamed into a
//    double-buffered LDS region with global_load_async_to_lds_b128 (ASYNCcnt),
//    hidden behind the WMMA work of the previous tile.
// Outputs (flat f32): [loss(1) | z_q_st(16*4096*512) | idx(65536) | perp(1)]
// ---------------------------------------------------------------------------

typedef __attribute__((ext_vector_type(16))) __bf16          v16bf;
typedef __attribute__((ext_vector_type(8)))  float           v8f;
typedef __attribute__((ext_vector_type(8)))  unsigned short  u16x8;
typedef __attribute__((ext_vector_type(4)))  unsigned short  u16x4;

#define NROWS    65536
#define EDIM     512
#define NCODES   1024
#define BM       64          // z rows per workgroup
#define NSTEPS   32          // 32 codes per step
#define LSTR     520         // padded LDS row stride (elems): 260 dw % 64 = 4 -> conflict-free b128
#define ZQ_ELEMS 33554432L   // 16*4096*512

// LDS partition (ushort units unless noted)
#define ZHI_U16   0                 // 64*520          = 33280
#define ZLO_U16   33280
#define CB_U16    66560             // 2 bufs * (hi 16640 + lo 16640) = 66560
#define EN_U16    133120            // 1024 f32 (4096 B)
#define REDV_B    270336            // 128 f32
#define REDI_B    270848            // 128 i32
#define LDS_BYTES 271360

__device__ __forceinline__ unsigned short f32_to_bf16(float x) {
  unsigned u = __float_as_uint(x);
  u += 0x7FFFu + ((u >> 16) & 1u);          // round-to-nearest-even
  return (unsigned short)(u >> 16);
}
__device__ __forceinline__ float bf16_to_f32(unsigned short h) {
  return __uint_as_float(((unsigned)h) << 16);
}
__device__ __forceinline__ int shfl_xor_i(int v, int m) {
  int lane = (int)(threadIdx.x & 31u);
  return __builtin_amdgcn_ds_bpermute(((lane ^ m) << 2), v);
}
__device__ __forceinline__ float shfl_xor_f(float v, int m) {
  return __int_as_float(shfl_xor_i(__float_as_int(v), m));
}
__device__ __forceinline__ v16bf load_frag(const unsigned short* p0, const unsigned short* p1) {
  union { u16x8 h[2]; v16bf v; } u;
  u.h[0] = *(const u16x8*)p0;
  u.h[1] = *(const u16x8*)p1;
  return u.v;
}
__device__ __forceinline__ void wait_async0() {
  asm volatile("s_wait_asynccnt 0x0" ::: "memory");
}

// Async-stage one 32-code tile (bf16 hi+lo) into LDS. Wave wv copies rows
// wv*4..wv*4+3; each instr moves 512 B (32 lanes x 16 B). IOFFSET is added to
// BOTH the LDS and global addresses, so the two row-halves share an addr pair.
__device__ __forceinline__ void issue_cb_tile(const unsigned short* __restrict__ gh,
                                              const unsigned short* __restrict__ gl,
                                              unsigned lds_hi_base, int step,
                                              int wv, int lane) {
  unsigned goff = (unsigned)((step * 32 + wv * 4) * 1024) + (unsigned)(lane * 16);
  unsigned dhi  = lds_hi_base + (unsigned)(wv * 4 * 1040) + (unsigned)(lane * 16);
  unsigned dlo  = dhi + 33280u;   // lo array sits 16640 ushorts after hi
#pragma unroll
  for (int r = 0; r < 4; ++r) {
    asm volatile(
        "global_load_async_to_lds_b128 %0, %1, %2\n\t"
        "global_load_async_to_lds_b128 %0, %1, %2 offset:512\n\t"
        "global_load_async_to_lds_b128 %3, %1, %4\n\t"
        "global_load_async_to_lds_b128 %3, %1, %4 offset:512"
        :
        : "v"(dhi), "v"(goff), "s"(gh), "v"(dlo), "s"(gl)
        : "memory");
    goff += 1024u;   // next codebook row (unpadded global)
    dhi  += 1040u;   // next LDS row (padded)
    dlo  += 1040u;
  }
}

// ---------------------------------------------------------------------------
// K_init: zero counts + loss accumulator
__global__ void k_init(float* __restrict__ counts, float* __restrict__ loss) {
  int t = threadIdx.x + blockIdx.x * blockDim.x;
  if (t < NCODES) counts[t] = 0.0f;
  if (t == 0) loss[0] = 0.0f;
}

// K_prep: split codebook f32 -> bf16 hi/lo in ws, and ||e||^2 per code.
// One wave per code, 8 codes per block.
__global__ __launch_bounds__(256) void k_prep(const float* __restrict__ cb,
                                              unsigned short* __restrict__ cbh,
                                              unsigned short* __restrict__ cbl,
                                              float* __restrict__ enorm) {
  int code = blockIdx.x * 8 + (threadIdx.x >> 5);
  int lane = threadIdx.x & 31;
  const float4* p = (const float4*)cb + (long)code * 128;
  float s = 0.0f;
#pragma unroll
  for (int i = 0; i < 4; ++i) {
    float4 v = p[lane * 4 + i];
    s += v.x * v.x + v.y * v.y + v.z * v.z + v.w * v.w;
    float xs[4] = {v.x, v.y, v.z, v.w};
    u16x4 h4, l4;
#pragma unroll
    for (int j = 0; j < 4; ++j) {
      unsigned short h = f32_to_bf16(xs[j]);
      h4[j] = h;
      l4[j] = f32_to_bf16(xs[j] - bf16_to_f32(h));
    }
    long o = (long)code * EDIM + lane * 16 + i * 4;
    *(u16x4*)(cbh + o) = h4;
    *(u16x4*)(cbl + o) = l4;
  }
#pragma unroll
  for (int m = 16; m >= 1; m >>= 1) s += shfl_xor_f(s, m);
  if (lane == 0) enorm[code] = s;
}

// ---------------------------------------------------------------------------
// K_dist: distances + argmin. grid = NROWS/64, block = 256 (8 waves).
// Waves: wm = wv>>1 picks 16-row group, wn = wv&1 picks 16-code half of tile.
__global__ __launch_bounds__(256) void k_dist(const float* __restrict__ z,
                                              const unsigned short* __restrict__ cbh,
                                              const unsigned short* __restrict__ cbl,
                                              const float* __restrict__ enorm_g,
                                              float* __restrict__ counts,
                                              float* __restrict__ out_idx_f,
                                              int* __restrict__ out_idx_i) {
  extern __shared__ unsigned short smem_u16[];
  unsigned short* zhi = smem_u16 + ZHI_U16;
  unsigned short* zlo = smem_u16 + ZLO_U16;
  unsigned short* cbt = smem_u16 + CB_U16;               // 2 x (hi|lo) tile bufs
  float* en_lds = (float*)(smem_u16 + EN_U16);
  float* redv = (float*)((char*)smem_u16 + REDV_B);
  int*   redi = (int*)((char*)smem_u16 + REDI_B);

  const int t = threadIdx.x;
  const long rowbase = (long)blockIdx.x * BM;
  const unsigned cb_lds_base = (unsigned)(size_t)(const void*)cbt;  // LDS byte addr

  // Stage 64x512 z tile, split f32 -> bf16 hi + residual lo
  for (int i = 0; i < 32; ++i) {
    int f4 = t + (i << 8);
    int r = f4 >> 7, c4 = f4 & 127;
    float4 v = ((const float4*)z)[(rowbase + r) * 128 + c4];
    float xs[4] = {v.x, v.y, v.z, v.w};
    u16x4 h4, l4;
#pragma unroll
    for (int j = 0; j < 4; ++j) {
      unsigned short h = f32_to_bf16(xs[j]);
      h4[j] = h;
      l4[j] = f32_to_bf16(xs[j] - bf16_to_f32(h));
    }
    *(u16x4*)(zhi + r * LSTR + (c4 << 2)) = h4;
    *(u16x4*)(zlo + r * LSTR + (c4 << 2)) = l4;
  }
  for (int i = t; i < NCODES; i += 256) en_lds[i] = enorm_g[i];

  const int lane = t & 31, wv = t >> 5;
  const int half = lane >> 4, mrow = lane & 15;
  const int wm = wv >> 1, wn = wv & 1;

  // A (16x32 bf16): chunks at K = 8*half + {0..7} and + {16..23}
  const unsigned short* a_h = zhi + (wm * 16 + mrow) * LSTR + half * 8;
  const unsigned short* a_l = zlo + (wm * 16 + mrow) * LSTR + half * 8;
  const int boff = (wn * 16 + mrow) * LSTR + half * 8;  // B: chunks at K = 8*half + {0..7} and + {16..23}

  // Prologue: async-stage tile 0 into buf 0
  issue_cb_tile(cbh, cbl, cb_lds_base, 0, wv, lane);
  wait_async0();
  __syncthreads();

  float minv[8];
  int   mini[8];
#pragma unroll
  for (int v = 0; v < 8; ++v) { minv[v] = 3.0e38f; mini[v] = 0; }

  for (int s = 0; s < NSTEPS; ++s) {
    // Prefetch next tile into the other buffer (hidden behind WMMAs below)
    if (s + 1 < NSTEPS)
      issue_cb_tile(cbh, cbl, cb_lds_base + ((unsigned)((s + 1) & 1)) * 66560u,
                    s + 1, wv, lane);

    const unsigned short* bb = cbt + (s & 1) * 33280;
    const unsigned short* b_h = bb + boff;
    const unsigned short* b_l = b_h + 16640;

    v8f acc0 = {}, acc1 = {}, acc2 = {};   // hi*hi, hi*lo, lo*hi
#pragma unroll
    for (int ks = 0; ks < 16; ++ks) {
      const int ao = ks * 32;
      v16bf ah  = load_frag(a_h + ao, a_h + ao + 16);
      v16bf al  = load_frag(a_l + ao, a_l + ao + 16);
      v16bf bh2 = load_frag(b_h + ao, b_h + ao + 16);
      v16bf bl2 = load_frag(b_l + ao, b_l + ao + 16);
      acc0 = __builtin_amdgcn_wmma_f32_16x16x32_bf16(false, ah, false, bh2, (short)0, acc0, false, false);
      acc1 = __builtin_amdgcn_wmma_f32_16x16x32_bf16(false, ah, false, bl2, (short)0, acc1, false, false);
      acc2 = __builtin_amdgcn_wmma_f32_16x16x32_bf16(false, al, false, bh2, (short)0, acc2, false, false);
    }

    const int code = s * 32 + wn * 16 + mrow;  // all 8 slots of a lane share one code
    const float en = en_lds[code];
#pragma unroll
    for (int v = 0; v < 8; ++v) {
      float dist = en - 2.0f * (acc0[v] + acc1[v] + acc2[v]);
      if (dist < minv[v]) { minv[v] = dist; mini[v] = code; }  // strict < keeps lowest idx
    }

    wait_async0();       // own async loads for tile s+1 complete
    __syncthreads();     // whole tile s+1 visible; buf (s&1) free for reuse
  }

  // Cross-lane argmin within each 16-lane column group
#pragma unroll
  for (int m = 1; m <= 8; m <<= 1) {
#pragma unroll
    for (int v = 0; v < 8; ++v) {
      float ov = shfl_xor_f(minv[v], m);
      int   oi = shfl_xor_i(mini[v], m);
      if (ov < minv[v] || (ov == minv[v] && oi < mini[v])) { minv[v] = ov; mini[v] = oi; }
    }
  }

  // Combine the two N-half waves per row group via LDS
  if (mrow == 0) {
#pragma unroll
    for (int v = 0; v < 8; ++v) {
      int e = (((wm * 2 + half) * 8) + v) * 2 + wn;
      redv[e] = minv[v];
      redi[e] = mini[v];
    }
  }
  __syncthreads();
  if (mrow == 0 && wn == 0) {
    long r0 = rowbase + wm * 16 + half * 8;
#pragma unroll
    for (int v = 0; v < 8; ++v) {
      int e0 = (((wm * 2 + half) * 8) + v) * 2;
      float v0 = redv[e0], v1 = redv[e0 + 1];
      int   i0 = redi[e0], i1 = redi[e0 + 1];
      bool take1 = (v1 < v0) || (v1 == v0 && i1 < i0);
      int bi = take1 ? i1 : i0;
      long row = r0 + v;
      out_idx_i[row] = bi;
      out_idx_f[row] = (float)bi;
      atomicAdd(&counts[bi], 1.0f);
    }
  }
}

// ---------------------------------------------------------------------------
// K_gather: z_q_st = codebook[idx]; accumulate sum((z_q - z)^2). 4 rows/block.
__global__ __launch_bounds__(256) void k_gather(const float* __restrict__ z,
                                                const float* __restrict__ cb,
                                                const int* __restrict__ idx,
                                                float* __restrict__ zq,   // +1 float misaligned -> scalar stores
                                                float* __restrict__ loss_acc) {
  __shared__ float red[8];
  long base = (long)blockIdx.x * 4;
  float part = 0.0f;
#pragma unroll
  for (int i = 0; i < 2; ++i) {
    int f4 = (int)threadIdx.x + (i << 8);
    int r = f4 >> 7, c4 = f4 & 127;
    long row = base + r;
    int code = idx[row];
    float4 e  = ((const float4*)cb)[(long)code * 128 + c4];
    float4 zz = ((const float4*)z)[row * 128 + c4];
    long o = row * EDIM + (long)c4 * 4;
    zq[o] = e.x; zq[o + 1] = e.y; zq[o + 2] = e.z; zq[o + 3] = e.w;
    float dx = e.x - zz.x, dy = e.y - zz.y, dz = e.z - zz.z, dw = e.w - zz.w;
    part += dx * dx + dy * dy + dz * dz + dw * dw;
  }
#pragma unroll
  for (int m = 16; m >= 1; m >>= 1) part += shfl_xor_f(part, m);
  if ((threadIdx.x & 31) == 0) red[threadIdx.x >> 5] = part;
  __syncthreads();
  if (threadIdx.x == 0) {
    float s = 0.0f;
#pragma unroll
    for (int i = 0; i < 8; ++i) s += red[i];
    atomicAdd(loss_acc, s);
  }
}

// K_final: loss = 1.25 * sum/len; perplexity = exp(-sum e_mean*log(e_mean+1e-10))
__global__ __launch_bounds__(1024) void k_final(const float* __restrict__ counts,
                                                const float* __restrict__ loss_acc,
                                                float* __restrict__ out_loss,
                                                float* __restrict__ out_perp) {
  __shared__ float red[32];
  int t = threadIdx.x;
  float c = counts[t] * (1.0f / 65536.0f);
  float term = c * logf(c + 1e-10f);
#pragma unroll
  for (int m = 16; m >= 1; m >>= 1) term += shfl_xor_f(term, m);
  if ((t & 31) == 0) red[t >> 5] = term;
  __syncthreads();
  if (t == 0) {
    float s = 0.0f;
#pragma unroll
    for (int i = 0; i < 32; ++i) s += red[i];
    out_perp[0] = expf(-s);
    out_loss[0] = 1.25f * loss_acc[0] * (1.0f / (float)ZQ_ELEMS);
  }
}

// ---------------------------------------------------------------------------
extern "C" void kernel_launch(void* const* d_in, const int* in_sizes, int n_in,
                              void* d_out, int out_size, void* d_ws, size_t ws_size,
                              hipStream_t stream) {
  const float* z  = (const float*)d_in[0];
  const float* cb = (const float*)d_in[1];

  float* out      = (float*)d_out;
  float* out_loss = out;
  float* out_zq   = out + 1;
  float* out_idx  = out + 1 + ZQ_ELEMS;
  float* out_perp = out_idx + NROWS;

  float* ws_enorm  = (float*)d_ws;                         // 1024 f32
  float* ws_counts = ws_enorm + NCODES;                    // 1024 f32
  float* ws_loss   = ws_counts + NCODES;                   // 1 f32 (+63 pad)
  int*   ws_idx    = (int*)(ws_loss + 64);                 // 65536 i32
  unsigned short* ws_cbh = (unsigned short*)(ws_idx + NROWS);   // 1024*512 bf16
  unsigned short* ws_cbl = ws_cbh + (long)NCODES * EDIM;        // 1024*512 bf16

  k_init<<<4, 256, 0, stream>>>(ws_counts, ws_loss);
  k_prep<<<NCODES / 8, 256, 0, stream>>>(cb, ws_cbh, ws_cbl, ws_enorm);
  k_dist<<<NROWS / BM, 256, LDS_BYTES, stream>>>(z, ws_cbh, ws_cbl, ws_enorm,
                                                 ws_counts, out_idx, ws_idx);
  k_gather<<<NROWS / 4, 256, 0, stream>>>(z, cb, ws_idx, out_zq, ws_loss);
  k_final<<<1, 1024, 0, stream>>>(ws_counts, ws_loss, out_loss, out_perp);
}
